// NeRFRenderer_75325136437502
// MI455X (gfx1250) — matrix operand order repaired and checked by reading, not compiled, not run on hardware
//
#include <hip/hip_runtime.h>

#define N_RAYS   16384
#define NSTEP    128
#define NUP      128
#define NTOT     256
#define HIDDEN   32
#define FEAT     15
#define IMGD     4
#define BOUNDF   2.0f
#define MIN_NEAR 0.05f

typedef __attribute__((ext_vector_type(16))) _Float16 v16h;
typedef __attribute__((ext_vector_type(8)))  float    v8f;

__device__ __forceinline__ v16h zero16h() {
    v16h r;
#pragma unroll
    for (int i = 0; i < 16; i++) r[i] = (_Float16)0.0f;
    return r;
}
__device__ __forceinline__ v8f zero8f() {
    v8f r;
#pragma unroll
    for (int i = 0; i < 8; i++) r[i] = 0.0f;
    return r;
}

union HU { unsigned u; _Float16 h[2]; };
__device__ __forceinline__ unsigned pack2(_Float16 a, _Float16 b) {
    HU x; x.h[0] = a; x.h[1] = b; return x.u;
}

__device__ __forceinline__ float softplus_f(float x) {
    return (x > 20.0f) ? x : log1pf(expf(x));
}

__device__ __forceinline__ void ray_setup(const float* __restrict__ ro,
                                          const float* __restrict__ rd, int ray,
                                          float& ox, float& oy, float& oz,
                                          float& dx, float& dy, float& dz,
                                          float& nr, float& fr) {
    ox = ro[3 * ray + 0]; oy = ro[3 * ray + 1]; oz = ro[3 * ray + 2];
    dx = rd[3 * ray + 0]; dy = rd[3 * ray + 1]; dz = rd[3 * ray + 2];
    float tmin = -3.0e38f, tmax = 3.0e38f;
    {
        float dv = (fabsf(dx) < 1e-9f) ? 1e-9f : dx; float inv = 1.0f / dv;
        float t1 = (-BOUNDF - ox) * inv, t2 = (BOUNDF - ox) * inv;
        tmin = fmaxf(tmin, fminf(t1, t2)); tmax = fminf(tmax, fmaxf(t1, t2));
    }
    {
        float dv = (fabsf(dy) < 1e-9f) ? 1e-9f : dy; float inv = 1.0f / dv;
        float t1 = (-BOUNDF - oy) * inv, t2 = (BOUNDF - oy) * inv;
        tmin = fmaxf(tmin, fminf(t1, t2)); tmax = fminf(tmax, fmaxf(t1, t2));
    }
    {
        float dv = (fabsf(dz) < 1e-9f) ? 1e-9f : dz; float inv = 1.0f / dv;
        float t1 = (-BOUNDF - oz) * inv, t2 = (BOUNDF - oz) * inv;
        tmin = fmaxf(tmin, fminf(t1, t2)); tmax = fminf(tmax, fmaxf(t1, t2));
    }
    nr = fmaxf(tmin, MIN_NEAR);
    fr = fmaxf(tmax, nr + 1e-5f);
}

// Build B-column tile for layer-1: lane < 16 holds its sample's xyz in K=0..2.
__device__ __forceinline__ v16h make_xyz_B(int lane, float z,
                                           float ox, float oy, float oz,
                                           float dx, float dy, float dz) {
    v16h bx = zero16h();
    if (lane < 16) {
        float px = fminf(fmaxf(ox + dx * z, -BOUNDF), BOUNDF);
        float py = fminf(fmaxf(oy + dy * z, -BOUNDF), BOUNDF);
        float pz = fminf(fmaxf(oz + dz * z, -BOUNDF), BOUNDF);
        bx[0] = (_Float16)px; bx[1] = (_Float16)py; bx[2] = (_Float16)pz;
    }
    return bx;
}

// Two-layer MLP for one 16-sample tile, weights-as-A orientation.
// Output: D2 channel-major (lane = sample (lane&15), VGPR r = channel Mb+r).
__device__ __forceinline__ v8f mlp_tile(int lane, int Mb, const v16h& aw0,
                                        const v16h& aw1, const v16h& a2,
                                        const float* __restrict__ b1,
                                        const v16h& bx) {
    v8f d0 = zero8f(), d1 = zero8f();
    d0 = __builtin_amdgcn_wmma_f32_16x16x32_f16(false, aw0, false, bx, (short)0, d0, false, false);
    d1 = __builtin_amdgcn_wmma_f32_16x16x32_f16(false, aw1, false, bx, (short)0, d1, false, false);

    // bias + ReLU, pack to f16 pairs
    unsigned p0[4], p1[4];
#pragma unroll
    for (int i = 0; i < 4; i++) {
        float h0a = fmaxf(d0[2 * i]     + b1[Mb + 2 * i],          0.0f);
        float h0b = fmaxf(d0[2 * i + 1] + b1[Mb + 2 * i + 1],      0.0f);
        float h1a = fmaxf(d1[2 * i]     + b1[16 + Mb + 2 * i],     0.0f);
        float h1b = fmaxf(d1[2 * i + 1] + b1[16 + Mb + 2 * i + 1], 0.0f);
        p0[i] = pack2((_Float16)h0a, (_Float16)h0b);
        p1[i] = pack2((_Float16)h1a, (_Float16)h1b);
    }
    // half-wave exchange: partner lane l^16 holds the other 8 hidden rows
    unsigned s0[4], s1[4];
#pragma unroll
    for (int i = 0; i < 4; i++) {
        s0[i] = (unsigned)__shfl_xor((int)p0[i], 16, 32);
        s1[i] = (unsigned)__shfl_xor((int)p1[i], 16, 32);
    }
    // assemble B2 = h^T: lane<16 -> K=0..15 (p0 own | p0 partner);
    //                    lane>=16 -> K=16..31 (p1 partner | p1 own)
    const bool lowh = (lane < 16);
    v16h b2v;
#pragma unroll
    for (int i = 0; i < 4; i++) {
        HU lo, hi;
        lo.u = lowh ? p0[i] : s1[i];
        hi.u = lowh ? s0[i] : p1[i];
        b2v[2 * i]         = lo.h[0];
        b2v[2 * i + 1]     = lo.h[1];
        b2v[8 + 2 * i]     = hi.h[0];
        b2v[8 + 2 * i + 1] = hi.h[1];
    }
    v8f c2 = zero8f();
    c2 = __builtin_amdgcn_wmma_f32_16x16x32_f16(false, a2, false, b2v, (short)0, c2, false, false);
    return c2;
}

// Load layer-1 A tiles (W1^T, rows = hidden, K = xyz) and layer-2 A (W2^T).
__device__ __forceinline__ void load_weight_tiles(int lane,
                                                  const float* __restrict__ W1,
                                                  const float* __restrict__ W2,
                                                  v16h& aw0, v16h& aw1, v16h& a2) {
    aw0 = zero16h(); aw1 = zero16h();
    if (lane < 16) {
#pragma unroll
        for (int k = 0; k < 3; k++) {
            aw0[k] = (_Float16)W1[k * HIDDEN + lane];
            aw1[k] = (_Float16)W1[k * HIDDEN + 16 + lane];
        }
    }
    {
        const int col = lane & 15;
        const int kb  = (lane < 16) ? 0 : 16;
#pragma unroll
        for (int j = 0; j < 8; j++) {
            a2[j]     = (_Float16)W2[(kb + j) * 16 + col];
            a2[8 + j] = (_Float16)W2[(kb + 8 + j) * 16 + col];
        }
    }
}

// ---------------------------------------------------------------------------
// Kernel 1: coarse sigma. 8 waves/block, one 16-sample tile per wave.
// No LDS, no barriers; sigma store is a coalesced 16-lane write.
// ---------------------------------------------------------------------------
__global__ __launch_bounds__(256) void nerf_coarse(
    const float* __restrict__ ro, const float* __restrict__ rd,
    const float* __restrict__ W1, const float* __restrict__ b1,
    const float* __restrict__ W2, const float* __restrict__ b2,
    float* __restrict__ sigma_out) {
    const int wave = threadIdx.x >> 5;
    const int lane = threadIdx.x & 31;
    const int tile = blockIdx.x * 8 + wave;
    const int ray  = tile >> 3;
    const int tloc = tile & 7;

    float ox, oy, oz, dx, dy, dz, nr, fr;
    ray_setup(ro, rd, ray, ox, oy, oz, dx, dy, dz, nr, fr);

    v16h aw0, aw1, a2;
    load_weight_tiles(lane, W1, W2, aw0, aw1, a2);

    float z = 0.0f;
    if (lane < 16) {
        int s = tloc * 16 + lane;
        z = nr + (fr - nr) * ((float)s * (1.0f / 127.0f));
    }
    v16h bx = make_xyz_B(lane, z, ox, oy, oz, dx, dy, dz);

    const int Mb = (lane >> 4) * 8;
    v8f c2 = mlp_tile(lane, Mb, aw0, aw1, a2, b1, bx);

    // channel 0 (sigma) lives in VGPR 0 of lanes 0..15 (channel = Mb + r)
    if (lane < 16)
        sigma_out[ray * NSTEP + tloc * 16 + lane] = softplus_f(c2[0] + b2[0]);
}

// ---------------------------------------------------------------------------
// Kernel 2: per-ray inverse-CDF resampling. One 128-thread block per ray.
// ---------------------------------------------------------------------------
__global__ __launch_bounds__(128) void nerf_resample(
    const float* __restrict__ ro, const float* __restrict__ rd,
    const float* __restrict__ sigma_in, float* __restrict__ newz) {
    __shared__ float zv[NSTEP];
    __shared__ float wbuf[NSTEP];
    __shared__ float cdf[NSTEP - 1];   // 127
    __shared__ float zmid[NSTEP - 1];  // 127

    const int ray = blockIdx.x;
    const int t   = threadIdx.x;

    float ox, oy, oz, dx, dy, dz, nr, fr;
    ray_setup(ro, rd, ray, ox, oy, oz, dx, dy, dz, nr, fr);

    zv[t] = nr + (fr - nr) * ((float)t * (1.0f / 127.0f));
    __syncthreads();

    if (t == 0) {
        const float sample_dist = (fr - nr) * (1.0f / 128.0f);
        float T = 1.0f;
        for (int i = 0; i < NSTEP; i++) {
            float delta = (i < NSTEP - 1) ? (zv[i + 1] - zv[i]) : sample_dist;
            float alpha = 1.0f - expf(-delta * sigma_in[ray * NSTEP + i]);
            wbuf[i] = alpha * T;
            T *= (1.0f - alpha + 1e-15f);
        }
        float s = 0.0f;
        for (int i = 1; i <= 126; i++) s += wbuf[i] + 1e-5f;
        float inv = 1.0f / s, acc = 0.0f;
        cdf[0] = 0.0f;
        for (int i = 1; i <= 126; i++) { acc += (wbuf[i] + 1e-5f) * inv; cdf[i] = acc; }
        for (int i = 0; i < 127; i++) zmid[i] = 0.5f * (zv[i] + zv[i + 1]);
    }
    __syncthreads();

    const float ulo = 0.5f / 128.0f, uhi = 1.0f - 0.5f / 128.0f;
    const float u = ulo + (uhi - ulo) * ((float)t * (1.0f / 127.0f));
    int l = 0, h = 127;  // searchsorted(cdf, u, 'right') over length-127 cdf
    while (l < h) { int m = (l + h) >> 1; if (cdf[m] <= u) l = m + 1; else h = m; }
    int below = min(max(l - 1, 0), 126);
    int above = min(l, 126);
    float cb = cdf[below], ca = cdf[above];
    float bb = zmid[below], ba = zmid[above];
    float den = ca - cb; if (den < 1e-5f) den = 1.0f;
    newz[ray * NUP + t] = bb + (u - cb) / den * (ba - bb);
}

// ---------------------------------------------------------------------------
// Kernel 3: merge + full MLP (WMMA) + color head + composite. One block/ray.
// ---------------------------------------------------------------------------
__global__ __launch_bounds__(256) void nerf_final(
    const float* __restrict__ ro, const float* __restrict__ rd,
    const float* __restrict__ W1, const float* __restrict__ b1,
    const float* __restrict__ W2, const float* __restrict__ b2,
    const float* __restrict__ Wc, const float* __restrict__ Wd,
    const float* __restrict__ bc,
    const float* __restrict__ newz, float* __restrict__ out) {
    __shared__ float zc[NSTEP];
    __shared__ float zf[NUP];
    __shared__ float zall[NTOT];
    __shared__ float outb[NTOT * 16];
    __shared__ float sg[NTOT];
    __shared__ float rgbb[NTOT * IMGD];

    const int ray = blockIdx.x;
    const int t   = threadIdx.x;

    float ox, oy, oz, dx, dy, dz, nr, fr;
    ray_setup(ro, rd, ray, ox, oy, oz, dx, dy, dz, nr, fr);

    if (t < NSTEP) zc[t] = nr + (fr - nr) * ((float)t * (1.0f / 127.0f));
    else           zf[t - NSTEP] = newz[ray * NUP + (t - NSTEP)];
    __syncthreads();

    // merge-path: both lists sorted; ranks replicate stable argsort of [zc, zf]
    {
        float myz; int rank;
        if (t < NSTEP) {
            myz = zc[t];
            int lo = 0, hi = NUP;
            while (lo < hi) { int m = (lo + hi) >> 1; if (zf[m] < myz) lo = m + 1; else hi = m; }
            rank = t + lo;
        } else {
            int j = t - NSTEP; myz = zf[j];
            int lo = 0, hi = NSTEP;
            while (lo < hi) { int m = (lo + hi) >> 1; if (zc[m] <= myz) lo = m + 1; else hi = m; }
            rank = j + lo;
        }
        zall[rank] = myz;
    }
    __syncthreads();

    const int wave = t >> 5, lane = t & 31;
    const int Mb = (lane >> 4) * 8;

    v16h aw0, aw1, a2;
    load_weight_tiles(lane, W1, W2, aw0, aw1, a2);

    // 16 tiles of 16 samples; 8 waves x 2 iterations, no barriers in the loop
#pragma unroll
    for (int it = 0; it < 2; ++it) {
        const int sbase = (wave * 2 + it) * 16;
        float z = (lane < 16) ? zall[sbase + lane] : 0.0f;
        v16h bx = make_xyz_B(lane, z, ox, oy, oz, dx, dy, dz);
        v8f c2 = mlp_tile(lane, Mb, aw0, aw1, a2, b1, bx);
        // channel-major D: lane owns 8 contiguous channels of its sample
        const int srow = sbase + (lane & 15);
#pragma unroll
        for (int r = 0; r < 8; r++)
            outb[srow * 16 + Mb + r] = c2[r] + b2[Mb + r];
    }
    __syncthreads();

    // per-sample sigma + color head
    {
        const float* op = &outb[t * 16];
        sg[t] = softplus_f(op[0]);
#pragma unroll
        for (int c = 0; c < IMGD; c++) {
            float acc = bc[c] + dx * Wd[0 * IMGD + c] + dy * Wd[1 * IMGD + c] + dz * Wd[2 * IMGD + c];
#pragma unroll
            for (int f = 0; f < FEAT; f++) acc += op[1 + f] * Wc[f * IMGD + c];
            rgbb[t * IMGD + c] = 1.0f / (1.0f + expf(-acc));
        }
    }
    __syncthreads();

    // serial alpha compositing (cumprod is inherently sequential)
    if (t == 0) {
        const float sample_dist = (fr - nr) * (1.0f / 128.0f);
        const float inv_span = 1.0f / (fr - nr);
        float T = 1.0f, depth = 0.0f, wsum = 0.0f;
        float img0 = 0.0f, img1 = 0.0f, img2 = 0.0f, img3 = 0.0f;
        for (int i = 0; i < NTOT; i++) {
            float delta = (i < NTOT - 1) ? (zall[i + 1] - zall[i]) : sample_dist;
            float alpha = 1.0f - expf(-delta * sg[i]);
            float w = alpha * T;
            T *= (1.0f - alpha + 1e-15f);
            float oz2 = fminf(fmaxf((zall[i] - nr) * inv_span, 0.0f), 1.0f);
            depth += w * oz2;
            wsum  += w;
            img0 += w * rgbb[i * IMGD + 0];
            img1 += w * rgbb[i * IMGD + 1];
            img2 += w * rgbb[i * IMGD + 2];
            img3 += w * rgbb[i * IMGD + 3];
        }
        float bg = 1.0f - wsum;  // bg_color = 1
        out[ray * IMGD + 0] = img0 + bg;
        out[ray * IMGD + 1] = img1 + bg;
        out[ray * IMGD + 2] = img2 + bg;
        out[ray * IMGD + 3] = img3 + bg;
        out[N_RAYS * IMGD + ray] = depth;
        out[N_RAYS * IMGD + N_RAYS + ray] = wsum;
    }
}

extern "C" void kernel_launch(void* const* d_in, const int* in_sizes, int n_in,
                              void* d_out, int out_size, void* d_ws, size_t ws_size,
                              hipStream_t stream) {
    (void)in_sizes; (void)n_in; (void)out_size; (void)ws_size;
    const float* ro = (const float*)d_in[0];
    const float* rd = (const float*)d_in[1];
    const float* W1 = (const float*)d_in[2];
    const float* b1 = (const float*)d_in[3];
    const float* W2 = (const float*)d_in[4];
    const float* b2 = (const float*)d_in[5];
    const float* Wc = (const float*)d_in[6];
    const float* Wd = (const float*)d_in[7];
    const float* bc = (const float*)d_in[8];
    float* out = (float*)d_out;

    float* sigma_c = (float*)d_ws;                       // 16384*128 floats
    float* newz    = sigma_c + (size_t)N_RAYS * NSTEP;   // 16384*128 floats

    nerf_coarse<<<N_RAYS, 256, 0, stream>>>(ro, rd, W1, b1, W2, b2, sigma_c);
    nerf_resample<<<N_RAYS, 128, 0, stream>>>(ro, rd, sigma_c, newz);
    nerf_final<<<N_RAYS, 256, 0, stream>>>(ro, rd, W1, b1, W2, b2, Wc, Wd, bc, newz, out);
}
